// AGCN_63136019251500
// MI455X (gfx1250) — compile-verified, matrix operand order checked by reading, not run on hardware
//
#include <hip/hip_runtime.h>
#include <hip/hip_bf16.h>

typedef __attribute__((ext_vector_type(16))) __bf16 bf16x16;
typedef __attribute__((ext_vector_type(8)))  float  f32x8;
typedef __attribute__((ext_vector_type(4)))  unsigned int u32x4;
typedef __attribute__((ext_vector_type(8)))  int i32x8;
typedef __attribute__((ext_vector_type(4)))  int i32x4;

#define WMMA_BF16(a, b, c) \
    __builtin_amdgcn_wmma_f32_16x16x32_bf16(false, (a), false, (b), (short)0, (c), false, false)

#if defined(__gfx1250__) && __has_builtin(__builtin_amdgcn_tensor_load_to_lds) && \
    __has_builtin(__builtin_amdgcn_s_wait_tensorcnt)
#define HAS_TDM 1
#else
#define HAS_TDM 0
#endif

// ---------------------------------------------------------------------------
// TDM: async DMA of one x tile (25 rows x 64 f32, contiguous) into LDS.
// D# per CDNA5 ISA: group0 = {count=1 | lds_addr |
// global_addr[56:0] | type=2}; group1 = {data_size=4B, tensor 64x25,
// tile 64x25, dim0 stride 64}. groups 2/3 unused (2D tensor) -> zero-filled.
// clang-23 (amdgpu-toolchain) signature: 6 args, trailing int32x8 + cpol.
// ---------------------------------------------------------------------------
#if HAS_TDM
__device__ __forceinline__ void tdm_issue_x(const float* gsrc, float* ldsDst) {
    unsigned long long ga = (unsigned long long)gsrc;
    unsigned lds = (unsigned)(unsigned long long)ldsDst;
    u32x4 g0;
    g0[0] = 1u;                                   // count=1 user descriptor
    g0[1] = lds;                                  // lds_addr (bytes)
    g0[2] = (unsigned)(ga & 0xFFFFFFFFu);         // global_addr[31:0]
    g0[3] = (unsigned)(ga >> 32) | (2u << 30);    // global_addr[56:32] | type=2
    i32x8 g1;
    g1[0] = 0x00020000;                           // data_size = 2 (4 bytes)
    g1[1] = (int)(64u << 16);                     // tensor_dim0 = 64
    g1[2] = (int)(25u << 16);                     // tensor_dim1 = 25
    g1[3] = (int)(64u << 16);                     // tile_dim0 = 64
    g1[4] = 25;                                   // tile_dim1 = 25
    g1[5] = 64;                                   // tensor_dim0_stride = 64
    g1[6] = 0; g1[7] = 0;
    i32x4 gz4 = {0, 0, 0, 0};
    i32x8 gz8 = {0, 0, 0, 0, 0, 0, 0, 0};
    __builtin_amdgcn_tensor_load_to_lds(g0, g1, gz4, gz4, gz8, 0);
}
#endif

// Issue stage: TDM from wave 0 only (TENSORcnt is per-wave), or synchronous
// fallback by all threads.
__device__ __forceinline__ void stage_x(const float* gsrc, float* ldsDst,
                                        int tid, int wid) {
#if HAS_TDM
    if (wid == 0) tdm_issue_x(gsrc, ldsDst);
    (void)tid;
#else
    for (int i = tid; i < 1600; i += 128) ldsDst[i] = gsrc[i];
    (void)wid;
#endif
}
__device__ __forceinline__ void stage_x_wait1(int wid) {
#if HAS_TDM
    if (wid == 0) __builtin_amdgcn_s_wait_tensorcnt(1);
#else
    (void)wid;
#endif
}
__device__ __forceinline__ void stage_x_wait0(int wid) {
#if HAS_TDM
    if (wid == 0) __builtin_amdgcn_s_wait_tensorcnt(0);
#else
    (void)wid;
#endif
}

// ---------------------------------------------------------------------------
// WMMA fragment helpers (wave32, V_WMMA_F32_16X16X32_BF16).
// A operand from row-major [M][K] bf16 LDS; B operand staged as B^T so the
// same contiguous-K loader applies (lane = N%16).
// ---------------------------------------------------------------------------
__device__ __forceinline__ bf16x16 ldsFrag(const __bf16* base, int stride,
                                           int mBase, int kBase, int lane) {
    int m  = mBase + (lane & 15);
    int kh = (lane & 16) ? 8 : 0;
    const __bf16* row = base + m * stride + kBase;
    union { uint4 q; __bf16 h[8]; } lo, hi;
    lo.q = *(const uint4*)(row + kh);
    hi.q = *(const uint4*)(row + 16 + kh);
    bf16x16 f;
#pragma unroll
    for (int i = 0; i < 8; ++i) { f[i] = lo.h[i]; f[i + 8] = hi.h[i]; }
    return f;
}

// Same fragment, sourced from f32 LDS (TDM-landed tile) with on-the-fly cvt.
__device__ __forceinline__ bf16x16 ldsFragF32(const float* base, int stride,
                                              int mBase, int kBase, int lane) {
    int m  = mBase + (lane & 15);
    int kh = (lane & 16) ? 8 : 0;
    const float* row = base + m * stride + kBase;
    float4 a0 = *(const float4*)(row + kh);
    float4 a1 = *(const float4*)(row + kh + 4);
    float4 b0 = *(const float4*)(row + 16 + kh);
    float4 b1 = *(const float4*)(row + 16 + kh + 4);
    bf16x16 f;
    f[0] = (__bf16)a0.x;  f[1] = (__bf16)a0.y;  f[2] = (__bf16)a0.z;  f[3] = (__bf16)a0.w;
    f[4] = (__bf16)a1.x;  f[5] = (__bf16)a1.y;  f[6] = (__bf16)a1.z;  f[7] = (__bf16)a1.w;
    f[8] = (__bf16)b0.x;  f[9] = (__bf16)b0.y;  f[10] = (__bf16)b0.z; f[11] = (__bf16)b0.w;
    f[12] = (__bf16)b1.x; f[13] = (__bf16)b1.y; f[14] = (__bf16)b1.z; f[15] = (__bf16)b1.w;
    return f;
}

// C/D layout: VGPR r, lanes 0-15 -> (M=r, N=lane); lanes 16-31 -> (M=r+8, N=lane-16)
__device__ __forceinline__ void stFragF32(float* base, int stride, int mBase,
                                          int nBase, f32x8 c, int lane) {
    int n  = nBase + (lane & 15);
    int mo = mBase + ((lane & 16) ? 8 : 0);
#pragma unroll
    for (int r = 0; r < 8; ++r) base[(mo + r) * stride + n] = c[r];
}

__device__ __forceinline__ void stFragBf16(__bf16* base, int stride, int mBase,
                                           int nBase, f32x8 c, int lane) {
    int n  = nBase + (lane & 15);
    int mo = mBase + ((lane & 16) ? 8 : 0);
#pragma unroll
    for (int r = 0; r < 8; ++r) base[(mo + r) * stride + n] = (__bf16)c[r];
}

// ---------------------------------------------------------------------------
// Kernel 0: A_comb = w0*A + w1*softmax(PA) (zero-padded 3x32x32), bd summed.
// ---------------------------------------------------------------------------
__global__ __launch_bounds__(128) void agcn_prep(const float* __restrict__ A,
                                                 const float* __restrict__ PA,
                                                 const float* __restrict__ wA,
                                                 const float* __restrict__ bd,
                                                 float* __restrict__ acomb,
                                                 float* __restrict__ bdsum) {
    int tid = threadIdx.x;
    for (int i = tid; i < 3 * 32 * 32; i += 128) acomb[i] = 0.f;
    if (tid < 64) bdsum[tid] = bd[tid] + bd[64 + tid] + bd[128 + tid];
    __syncthreads();
    float w0 = wA[0], w1 = wA[1];
    if (tid < 75) {
        int s = tid / 25, u = tid % 25;
        const float* row  = PA + (s * 25 + u) * 25;
        const float* arow = A + (s * 25 + u) * 25;
        float m = -1e30f;
        for (int v = 0; v < 25; ++v) m = fmaxf(m, row[v]);
        float e[25], sum = 0.f;
        for (int v = 0; v < 25; ++v) { e[v] = __expf(row[v] - m); sum += e[v]; }
        float inv = w1 / sum;
        float* orow = acomb + s * 1024 + u * 32;
        for (int v = 0; v < 25; ++v) orow[v] = w0 * arow[v] + e[v] * inv;
    }
}

// ---------------------------------------------------------------------------
// Kernel 1: partial A1 reduction. Grid = S*N*4 chunks = 384 blocks; each block
// accumulates sum_t qa(t) @ kb(t)^T over 64 t-steps with TDM double-buffered
// x tiles, writes a raw 32x32 f32 partial to workspace.
// ---------------------------------------------------------------------------
__global__ __launch_bounds__(128) void agcn_a1_part(const float* __restrict__ x,
                                                    const float* __restrict__ Wa,
                                                    const float* __restrict__ ba,
                                                    const float* __restrict__ Wb,
                                                    const float* __restrict__ bb,
                                                    float* __restrict__ scpart) {
    __shared__ __align__(16) float  sxf[2][2048];
    __shared__ __align__(16) __bf16 sWaT[1024], sWbT[1024], sqa[1024], skb[1024];
    __shared__ float sba[16], sbb[16];

    int tid = threadIdx.x, lane = tid & 31, wid = tid >> 5;
    int sIdx = blockIdx.x >> 7, rem = blockIdx.x & 127, n = rem >> 2, ch = rem & 3;
    int t0 = ch * 64;

    stage_x(x + (size_t)((n * 256 + t0) * 25) * 64, sxf[0], tid, wid);
    for (int i = tid; i < 1024; i += 128) {
        int ic = i >> 6, c = i & 63;
        sWaT[i] = (__bf16)Wa[(sIdx * 64 + c) * 16 + ic];
        sWbT[i] = (__bf16)Wb[(sIdx * 64 + c) * 16 + ic];
        sqa[i] = (__bf16)0.f; skb[i] = (__bf16)0.f;
    }
    if (tid < 16) { sba[tid] = ba[sIdx * 16 + tid]; sbb[tid] = bb[sIdx * 16 + tid]; }
    for (int i = tid; i < 2 * 448; i += 128) {     // zero pad rows 25..31
        int b = (i >= 448) ? 1 : 0;
        sxf[b][1600 + (i - b * 448)] = 0.f;
    }

    int mi = wid >> 1, ni = wid & 1, buf = 0;
    f32x8 acc = {};
    for (int t = 0; t < 64; ++t) {
        if (t + 1 < 64) {
            stage_x(x + (size_t)((n * 256 + t0 + t + 1) * 25) * 64, sxf[buf ^ 1], tid, wid);
            stage_x_wait1(wid);
        } else {
            stage_x_wait0(wid);
        }
        __syncthreads();
        const float* sxb = sxf[buf];
        {   // waves 0/1: qa tiles 0/1 ; waves 2/3: kb tiles 0/1
            const __bf16* W    = (wid < 2) ? sWaT : sWbT;
            const float*  bias = (wid < 2) ? sba  : sbb;
            __bf16*       dst  = (wid < 2) ? sqa  : skb;
            int mB = (wid & 1) * 16;
            f32x8 p = {};
#pragma unroll
            for (int kk = 0; kk < 64; kk += 32) {
                bf16x16 a = ldsFragF32(sxb, 64, mB, kk, lane);
                bf16x16 b = ldsFrag(W, 64, 0, kk, lane);
                p = WMMA_BF16(a, b, p);
            }
            float bv = bias[lane & 15];
#pragma unroll
            for (int r = 0; r < 8; ++r) p[r] += bv;
            stFragBf16(dst, 32, mB, 0, p, lane);
        }
        __syncthreads();
        {   // score accumulation (K padded 16->32 with zeros)
            bf16x16 a = ldsFrag(sqa, 32, mi * 16, 0, lane);
            bf16x16 b = ldsFrag(skb, 32, ni * 16, 0, lane);
            acc = WMMA_BF16(a, b, acc);
        }
        __syncthreads();
        buf ^= 1;
    }
    stFragF32(scpart + (size_t)((sIdx * 32 + n) * 4 + ch) * 1024, 32,
              mi * 16, ni * 16, acc, lane);
}

// ---------------------------------------------------------------------------
// Kernel 1b: reduce 4 partials, softmax (/64), pre-scale by w2. Grid = 96.
// ---------------------------------------------------------------------------
__global__ __launch_bounds__(32) void agcn_a1_reduce(const float* __restrict__ wA,
                                                     const float* __restrict__ scpart,
                                                     float* __restrict__ a1w) {
    int sIdx = blockIdx.x >> 5, n = blockIdx.x & 31;
    int u = threadIdx.x;
    const float* base = scpart + (size_t)((sIdx * 32 + n) * 4) * 1024;
    float* orow = a1w + (size_t)(sIdx * 32 + n) * 1024 + u * 32;
    if (u < 25) {
        float w2 = wA[2];
        float r[25], m = -1e30f;
        for (int v = 0; v < 25; ++v) {
            int o = u * 32 + v;
            r[v] = (base[o] + base[1024 + o] + base[2048 + o] + base[3072 + o]) * (1.f / 64.f);
            m = fmaxf(m, r[v]);
        }
        float sum = 0.f;
        for (int v = 0; v < 25; ++v) { r[v] = __expf(r[v] - m); sum += r[v]; }
        float sc = w2 / sum;
        for (int v = 0; v < 25; ++v) orow[v] = r[v] * sc;
        for (int v = 25; v < 32; ++v) orow[v] = 0.f;
    } else {
        for (int v = 0; v < 32; ++v) orow[v] = 0.f;
    }
}

// ---------------------------------------------------------------------------
// Kernel 2: fused main pass. Grid = N*T/4 = 2048 blocks; each block stages
// weights once and processes 4 consecutive t-tiles with TDM double buffering.
// ---------------------------------------------------------------------------
__global__ __launch_bounds__(128) void agcn_main(
    const float* __restrict__ x, const float* __restrict__ wA,
    const float* __restrict__ Wa, const float* __restrict__ ba,
    const float* __restrict__ Wb, const float* __restrict__ bb,
    const float* __restrict__ Wd, const float* __restrict__ gamma,
    const float* __restrict__ beta, const float* __restrict__ acomb,
    const float* __restrict__ bdsum, const float* __restrict__ a1w,
    float* __restrict__ out) {
    __shared__ __align__(16) float  sxf[2][2048];
    __shared__ __align__(16) __bf16 sWaT[3][1024], sWbT[3][1024], sWdT[3][4096];
    __shared__ __align__(16) __bf16 sx[2048], sxT[2048];
    __shared__ __align__(16) __bf16 sqa[1024], skb[1024], sain[1024], sz[2048];
    __shared__ float ssc[1024];
    __shared__ float sAc[3][1024], sA1[3][1024];
    __shared__ float sba[3][16], sbb[3][16], sbd[64];

    int tid = threadIdx.x, lane = tid & 31, wid = tid >> 5;
    int n = blockIdx.x >> 6, tBase = (blockIdx.x & 63) * 4;

    stage_x(x + (size_t)((n * 256 + tBase) * 25) * 64, sxf[0], tid, wid);
    for (int i = tid; i < 3 * 1024; i += 128) {
        int s = i >> 10, r = i & 1023, ic = r >> 6, c = r & 63;
        sWaT[s][r] = (__bf16)Wa[(s * 64 + c) * 16 + ic];
        sWbT[s][r] = (__bf16)Wb[(s * 64 + c) * 16 + ic];
        sAc[s][r]  = acomb[i];
        sA1[s][r]  = a1w[(size_t)(s * 32 + n) * 1024 + r];
    }
    for (int i = tid; i < 3 * 4096; i += 128) {
        int s = i >> 12, r = i & 4095, o = r >> 6, c = r & 63;
        sWdT[s][r] = (__bf16)Wd[(s * 64 + c) * 64 + o];
    }
    if (tid < 48) { sba[tid >> 4][tid & 15] = ba[tid]; sbb[tid >> 4][tid & 15] = bb[tid]; }
    if (tid < 64) sbd[tid] = bdsum[tid];
    for (int i = tid; i < 1024; i += 128) { sqa[i] = (__bf16)0.f; skb[i] = (__bf16)0.f; }
    for (int i = tid; i < 2 * 448; i += 128) {     // zero pad rows 25..31
        int b = (i >= 448) ? 1 : 0;
        sxf[b][1600 + (i - b * 448)] = 0.f;
    }

    float w3 = wA[3];
    int ni = wid, buf = 0;
    for (int tt = 0; tt < 4; ++tt) {
        int t = tBase + tt;
        if (tt + 1 < 4) {
            stage_x(x + (size_t)((n * 256 + t + 1) * 25) * 64, sxf[buf ^ 1], tid, wid);
            stage_x_wait1(wid);
        } else {
            stage_x_wait0(wid);
        }
        __syncthreads();
        const float* sxb = sxf[buf];
        for (int i = tid; i < 2048; i += 128) {    // cvt to bf16 (pad rows are 0)
            int r = i >> 6, c = i & 63;
            __bf16 v = (__bf16)sxb[i];
            sx[i] = v;
            sxT[c * 32 + r] = v;
        }
        __syncthreads();

        f32x8 y0 = {}, y1 = {};
        for (int s = 0; s < 3; ++s) {
            {   // qa/kb projections + bias
                const __bf16* W    = (wid < 2) ? sWaT[s] : sWbT[s];
                const float*  bias = (wid < 2) ? sba[s]  : sbb[s];
                __bf16*       dst  = (wid < 2) ? sqa     : skb;
                int mB = (wid & 1) * 16;
                f32x8 p = {};
#pragma unroll
                for (int kk = 0; kk < 64; kk += 32) {
                    bf16x16 a = ldsFrag(sx, 64, mB, kk, lane);
                    bf16x16 b = ldsFrag(W, 64, 0, kk, lane);
                    p = WMMA_BF16(a, b, p);
                }
                float bv = bias[lane & 15];
#pragma unroll
                for (int r = 0; r < 8; ++r) p[r] += bv;
                stFragBf16(dst, 32, mB, 0, p, lane);
            }
            __syncthreads();
            {   // scores = qa @ kb^T
                int smi = wid >> 1, sni = wid & 1;
                bf16x16 a = ldsFrag(sqa, 32, smi * 16, 0, lane);
                bf16x16 b = ldsFrag(skb, 32, sni * 16, 0, lane);
                f32x8 zero = {};
                f32x8 sc = WMMA_BF16(a, b, zero);
                stFragF32(ssc, 32, smi * 16, sni * 16, sc, lane);
            }
            __syncthreads();
            if (tid < 32) {  // A_in = A_comb + w2*A1 + w3*softmax(scores/4)
                int u = tid;
                __bf16* arow = sain + u * 32;
                if (u < 25) {
                    float m = -1e30f;
                    for (int v = 0; v < 25; ++v) m = fmaxf(m, ssc[u * 32 + v] * 0.25f);
                    float e[25], sum = 0.f;
                    for (int v = 0; v < 25; ++v) {
                        e[v] = __expf(ssc[u * 32 + v] * 0.25f - m);
                        sum += e[v];
                    }
                    float inv = w3 / sum;
                    for (int v = 0; v < 25; ++v)
                        arow[v] = (__bf16)(sAc[s][u * 32 + v] + sA1[s][u * 32 + v] + e[v] * inv);
                    for (int v = 25; v < 32; ++v) arow[v] = (__bf16)0.f;
                } else {
                    for (int v = 0; v < 32; ++v) arow[v] = (__bf16)0.f;
                }
            }
            __syncthreads();
            {   // z = A_in @ x (K=32), N tile = ni
                bf16x16 b = ldsFrag(sxT, 32, ni * 16, 0, lane);
#pragma unroll
                for (int mi2 = 0; mi2 < 2; ++mi2) {
                    bf16x16 a = ldsFrag(sain, 32, mi2 * 16, 0, lane);
                    f32x8 zero = {};
                    f32x8 z = WMMA_BF16(a, b, zero);
                    stFragBf16(sz, 64, mi2 * 16, ni * 16, z, lane);
                }
            }
            __syncthreads();
            {   // y += z @ Wd_s (K=64)
#pragma unroll
                for (int kk = 0; kk < 64; kk += 32) {
                    bf16x16 b  = ldsFrag(sWdT[s], 64, ni * 16, kk, lane);
                    bf16x16 a0 = ldsFrag(sz, 64, 0, kk, lane);
                    bf16x16 a1 = ldsFrag(sz, 64, 16, kk, lane);
                    y0 = WMMA_BF16(a0, b, y0);
                    y1 = WMMA_BF16(a1, b, y1);
                }
            }
            __syncthreads();
        }

        {   // epilogue: +bd_sum, BN, +beta, residual (from LDS f32 x), relu
            int o = ni * 16 + (lane & 15);
            float g = gamma[o], bt = beta[o], bs = sbd[o];
            float scale = g * rsqrtf(1.0f + 1e-5f);
            int mo = (lane & 16) ? 8 : 0;
#pragma unroll
            for (int r = 0; r < 8; ++r) {
                int m0 = mo + r;
                if (m0 < 25) {
                    size_t idx = (size_t)((n * 256 + t) * 25 + m0) * 64 + o;
                    float v = (y0[r] + bs) * scale + bt + sxb[m0 * 64 + o];
                    out[idx] = fmaxf(v, 0.f);
                }
                int m1 = 16 + mo + r;
                if (m1 < 25) {
                    size_t idx = (size_t)((n * 256 + t) * 25 + m1) * 64 + o;
                    float v = (y1[r] + bs) * scale + bt + sxb[m1 * 64 + o];
                    out[idx] = fmaxf(v, 0.f);
                }
            }
        }
        __syncthreads();   // protect sxf[buf]/sx/sxT before next tt's TDM+cvt
        buf ^= 1;
    }
}

extern "C" void kernel_launch(void* const* d_in, const int* in_sizes, int n_in,
                              void* d_out, int out_size, void* d_ws, size_t ws_size,
                              hipStream_t stream) {
    (void)in_sizes; (void)n_in; (void)out_size; (void)ws_size;
    const float* x     = (const float*)d_in[0];
    const float* A     = (const float*)d_in[1];
    const float* PA    = (const float*)d_in[2];
    const float* wA    = (const float*)d_in[3];
    const float* Wa    = (const float*)d_in[4];
    const float* ba    = (const float*)d_in[5];
    const float* Wb    = (const float*)d_in[6];
    const float* bb    = (const float*)d_in[7];
    const float* Wd    = (const float*)d_in[8];
    const float* bd    = (const float*)d_in[9];
    const float* gamma = (const float*)d_in[10];
    const float* beta  = (const float*)d_in[11];

    float* ws     = (float*)d_ws;
    float* acomb  = ws;                 // 3*32*32            = 3072
    float* bdsum  = ws + 3072;          // 64
    float* a1w    = ws + 3136;          // 3*32*1024          = 98304
    float* scpart = ws + 101440;        // 3*32*4*1024        = 393216
    float* out    = (float*)d_out;

    agcn_prep<<<dim3(1), dim3(128), 0, stream>>>(A, PA, wA, bd, acomb, bdsum);
    agcn_a1_part<<<dim3(384), dim3(128), 0, stream>>>(x, Wa, ba, Wb, bb, scpart);
    agcn_a1_reduce<<<dim3(96), dim3(32), 0, stream>>>(wA, scpart, a1w);
    agcn_main<<<dim3(2048), dim3(128), 0, stream>>>(x, wA, Wa, ba, Wb, bb, Wd,
                                                    gamma, beta, acomb, bdsum,
                                                    a1w, out);
}
